// UnetrUpBlock2_61993557950805
// MI455X (gfx1250) — compile-verified, matrix operand order checked by reading, not run on hardware
//
#include <hip/hip_runtime.h>
#include <hip/hip_bf16.h>

typedef __bf16 bf16_t;
typedef __attribute__((ext_vector_type(16))) __bf16 v16bf;
typedef __attribute__((ext_vector_type(8)))  __bf16 v8bf;
typedef __attribute__((ext_vector_type(8)))  float  v8f;

#define NTOK  343      // tokens per 7^3 window
#define NPAD  352      // 343 padded to 22*16
#define DIMC  192
#define NHEAD 6
#define HD    32
#define NWIN  128      // B * 64 windows
#define SEQ   21952    // 28^3
#define MROWS 43904    // 2 * 21952  (divisible by 32)
#define FFN   768

// ---------------- wave helpers ----------------
__device__ __forceinline__ float wave_sum(float v) {
#pragma unroll
  for (int off = 16; off > 0; off >>= 1) v += __shfl_xor(v, off, 32);
  return v;
}
__device__ __forceinline__ float half_sum(float v) {   // reduce within 16-lane half
#pragma unroll
  for (int off = 1; off < 16; off <<= 1) v += __shfl_xor(v, off, 32);
  return v;
}
__device__ __forceinline__ float half_max(float v) {
#pragma unroll
  for (int off = 1; off < 16; off <<= 1) v = fmaxf(v, __shfl_xor(v, off, 32));
  return v;
}
__device__ __forceinline__ v8f zero_acc() {
  v8f a;
#pragma unroll
  for (int i = 0; i < 8; ++i) a[i] = 0.f;
  return a;
}

// ---------------- WMMA fragment / tile primitives ----------------
// 16-bit A-matrix 16x32 fragment layout (ISA 7.12.2):
//   lanes 0-15 : VGPR0-3 = K0..7,  VGPR4-7 = K16..23   (row M = lane)
//   lanes 16-31: VGPR0-3 = K8..15, VGPR4-7 = K24..31
// -> per lane: two contiguous 16B loads at k0+half*8 and k0+16+half*8.
// B is supplied as Bt (N-major, K contiguous), mirrored layout.
__device__ __forceinline__ v16bf load_frag_bf16(const bf16_t* __restrict__ base,
                                                int ld, int row, int k0, int half) {
  const bf16_t* p = base + (size_t)row * ld + k0 + half * 8;
  v8bf lo = *(const v8bf*)(p);
  v8bf hi = *(const v8bf*)(p + 16);
  v16bf f;
#pragma unroll
  for (int i = 0; i < 8; ++i) { f[i] = lo[i]; f[i + 8] = hi[i]; }
  return f;
}

// 2x2 register-blocked tile: 32x32 output per wave, 4 WMMAs per 4 fragment loads.
template <int BM, int BN>
__device__ __forceinline__ void gemm_block_bf16(const bf16_t* __restrict__ A, int lda,
                                                const bf16_t* __restrict__ Bt, int ldb,
                                                int K, int m0, int n0, v8f acc[BM][BN]) {
  const int lane = threadIdx.x & 31;
  const int half = lane >> 4;
  const int r    = lane & 15;
  for (int k0 = 0; k0 < K; k0 += 32) {
    if (k0 + 32 < K) {
      __builtin_prefetch(A  + (size_t)(m0 + r) * lda + k0 + 32, 0, 0);
      __builtin_prefetch(Bt + (size_t)(n0 + r) * ldb + k0 + 32, 0, 0);
    }
    v16bf a[BM], b[BN];
#pragma unroll
    for (int i = 0; i < BM; ++i) a[i] = load_frag_bf16(A,  lda, m0 + i * 16 + r, k0, half);
#pragma unroll
    for (int j = 0; j < BN; ++j) b[j] = load_frag_bf16(Bt, ldb, n0 + j * 16 + r, k0, half);
#pragma unroll
    for (int i = 0; i < BM; ++i)
#pragma unroll
      for (int j = 0; j < BN; ++j)
        acc[i][j] = __builtin_amdgcn_wmma_f32_16x16x32_bf16(false, a[i], false, b[j],
                                                            (short)0, acc[i][j], false, false);
  }
}
// C tile layout (ISA): elem e -> row m0+e+8*half, col n0+(lane&15)

// ---------------- weight transpose + bf16 convert ----------------
__global__ void k_transpose_w(const float* __restrict__ W, bf16_t* __restrict__ Wt,
                              int K, int Nc) {
  const int i = blockIdx.x * 256 + threadIdx.x;
  if (i >= K * Nc) return;
  const int k = i / Nc, n = i % Nc;
  Wt[(size_t)n * K + k] = (bf16_t)W[i];
}

// ---------------- LN1 + cyclic shift + window partition -> bf16 ----------------
__global__ void k_ln1_shift_partition(const float* __restrict__ x,
                                      const float* __restrict__ g,
                                      const float* __restrict__ bb,
                                      bf16_t* __restrict__ Wbuf) {
  const int wave = threadIdx.x >> 5, lane = threadIdx.x & 31;
  const int t = blockIdx.x * 8 + wave;
  if (t >= NPAD) return;
  const int win = blockIdx.y;
  bf16_t* dst = Wbuf + ((size_t)win * NPAD + t) * DIMC;
  if (t >= NTOK) {               // zero pad rows so WMMA reads are defined
#pragma unroll
    for (int i = 0; i < 6; ++i) dst[lane + 32 * i] = (bf16_t)0.f;
    return;
  }
  const int b  = win >> 6;
  const int w  = win & 63;
  const int sz = w >> 4, sy = (w >> 2) & 3, sx = w & 3;
  const int zi = t / 49, yi = (t / 7) % 7, xi = t % 7;
  const int z  = (sz * 7 + zi + 3) % 28;   // roll(-3)
  const int y  = (sy * 7 + yi + 3) % 28;
  const int xx = (sx * 7 + xi + 3) % 28;
  const int p  = (z * 28 + y) * 28 + xx;
  const float* src = x + ((size_t)b * SEQ + p) * DIMC;
  float v[6]; float s = 0.f;
#pragma unroll
  for (int i = 0; i < 6; ++i) { v[i] = src[lane + 32 * i]; s += v[i]; }
  const float mean = wave_sum(s) * (1.f / DIMC);
  float q = 0.f;
#pragma unroll
  for (int i = 0; i < 6; ++i) { const float d = v[i] - mean; q += d * d; }
  const float rstd = rsqrtf(wave_sum(q) * (1.f / DIMC) + 1e-5f);
#pragma unroll
  for (int i = 0; i < 6; ++i) {
    const int c = lane + 32 * i;
    dst[c] = (bf16_t)((v[i] - mean) * rstd * g[c] + bb[c]);
  }
}

// ---------------- QKV GEMM: [352x192] x [192x576], 32x32 per wave ----------------
__global__ void k_qkv_gemm(const bf16_t* __restrict__ Wbuf, const bf16_t* __restrict__ Bt,
                           const float* __restrict__ qkv_b,
                           bf16_t* __restrict__ qb, bf16_t* __restrict__ kb,
                           bf16_t* __restrict__ vT) {
  const int wave = threadIdx.x >> 5;
  const int tile = blockIdx.x * 8 + wave;
  const int TM = NPAD / 32, TN = 576 / 32;   // 11 x 18
  if (tile >= TM * TN) return;
  const int win = blockIdx.y;
  const int tm = tile / TN, tn = tile % TN;
  v8f acc[2][2] = {{zero_acc(), zero_acc()}, {zero_acc(), zero_acc()}};
  gemm_block_bf16<2, 2>(Wbuf + (size_t)win * NPAD * DIMC, DIMC, Bt, DIMC, DIMC,
                        tm * 32, tn * 32, acc);
  const int lane = threadIdx.x & 31, half = lane >> 4, r = lane & 15;
#pragma unroll
  for (int j = 0; j < 2; ++j) {
    const int col = tn * 32 + j * 16 + r;
    const float bias = qkv_b[col];
    const int seg = col / DIMC, cc = col % DIMC;
    const int h = (cc >> 5), d = cc & 31;
    const size_t bh = (size_t)win * NHEAD + h;
#pragma unroll
    for (int i = 0; i < 2; ++i)
#pragma unroll
      for (int e = 0; e < 8; ++e) {
        const int row = tm * 32 + i * 16 + e + 8 * half;
        const float val = acc[i][j][e] + bias;
        if (seg == 0)      qb[(bh * NPAD + row) * HD + d] = (bf16_t)(val * 0.17677669529663687f);
        else if (seg == 1) kb[(bh * NPAD + row) * HD + d] = (bf16_t)val;
        else               vT[(bh * HD + d) * NPAD + row] = (bf16_t)val;  // v transposed
      }
  }
}

// ---------------- flash attention ----------------
// All 8 waves of a block share (window, head): the 32-key K/V blocks are staged
// cooperatively into LDS with async global->LDS copies (one b128 per thread),
// then each wave runs S = q kT (2 WMMA), online softmax, probs via a private
// 1KB LDS tile re-shaped into an A-fragment, O += P V (2 WMMA).
__global__ void __launch_bounds__(256) k_flash_attn(
    const bf16_t* __restrict__ qb, const bf16_t* __restrict__ kb,
    const bf16_t* __restrict__ vT, const float* __restrict__ rel_table,
    const float* __restrict__ attn_mask, bf16_t* __restrict__ ob) {
  __shared__ __align__(16) bf16_t lds_p[8][16 * 32];   // per-wave probs staging
  __shared__ __align__(16) bf16_t lds_k[32 * 32];      // K block: 32 keys x 32 hd
  __shared__ __align__(16) bf16_t lds_v[32 * 32];      // V block: 32 hd  x 32 keys
  const int tid  = threadIdx.x;
  const int wave = tid >> 5, lane = tid & 31;
  const int half = lane >> 4, r = lane & 15;
  const int tm = blockIdx.x * 8 + wave;
  const bool active = (tm < NPAD / 16);
  const int tmc = active ? tm : 0;           // inactive waves compute tile 0, skip stores
  const int bh = blockIdx.y;                 // win*6 + head
  const int h = bh % NHEAD, win = bh / NHEAD, nw = win & 63;

  const bf16_t* Q = qb + (size_t)bh * NPAD * HD;
  const bf16_t* K = kb + (size_t)bh * NPAD * HD;
  const bf16_t* V = vT + (size_t)bh * HD * NPAD;

  // async-staging assignment: threads 0..127 copy K tile, 128..255 copy V tile.
  const int  srow = (tid & 127) >> 2;        // 0..31 rows
  const int  schk = (tid & 3) * 8;           // bf16 element offset 0/8/16/24 (16B chunks)
  const bool doK  = (tid < 128);
  const unsigned lds_dst = (unsigned)(uintptr_t)(doK ? &lds_k[srow * 32 + schk]
                                                     : &lds_v[srow * 32 + schk]);

  const v16bf aq = load_frag_bf16(Q, HD, tmc * 16 + r, 0, half);  // q loaded once

  v8f o0 = zero_acc(), o1 = zero_acc();
  float mrun[8], lrun[8];
  int rowc[8], rz[8], ry[8], rx[8];
#pragma unroll
  for (int e = 0; e < 8; ++e) {
    mrun[e] = -3.0e38f; lrun[e] = 0.f;
    const int row = tmc * 16 + e + 8 * half;
    const int rr = (row < NTOK) ? row : 0;   // clamp pad rows for table lookups
    rowc[e] = rr; rz[e] = rr / 49; ry[e] = (rr / 7) % 7; rx[e] = rr % 7;
  }
  bf16_t* pl = lds_p[wave];

  for (int key0 = 0; key0 < NPAD; key0 += 32) {
    __syncthreads();   // previous iteration's LDS reads done before overwrite
    const bf16_t* gsrc = doK ? (K + (size_t)(key0 + srow) * HD + schk)
                             : (V + (size_t)srow * NPAD + key0 + schk);
    asm volatile("global_load_async_to_lds_b128 %0, %1, off"
                 :: "v"(lds_dst), "v"(gsrc) : "memory");
    asm volatile("s_wait_asynccnt 0x0" ::: "memory");
    __syncthreads();

    const v16bf bk0 = load_frag_bf16(lds_k, 32, r, 0, half);
    const v16bf bk1 = load_frag_bf16(lds_k, 32, 16 + r, 0, half);
    v8f s0 = zero_acc(), s1 = zero_acc();
    s0 = __builtin_amdgcn_wmma_f32_16x16x32_bf16(false, aq, false, bk0, (short)0, s0, false, false);
    s1 = __builtin_amdgcn_wmma_f32_16x16x32_bf16(false, aq, false, bk1, (short)0, s1, false, false);

    const int c0 = key0 + r, c1 = key0 + 16 + r;
    const int c0z = c0 / 49, c0y = (c0 / 7) % 7, c0x = c0 % 7;
    const int c1z = c1 / 49, c1y = (c1 / 7) % 7, c1x = c1 % 7;
    float p0[8], p1[8];
#pragma unroll
    for (int e = 0; e < 8; ++e) {
      float v0 = -3.0e38f, v1 = -3.0e38f;
      if (c0 < NTOK) {
        const int idx = (rz[e] - c0z + 6) * 20 + (ry[e] - c0y + 6) * 13 + (rx[e] - c0x + 6);
        v0 = s0[e] + rel_table[idx * NHEAD + h]
                   + attn_mask[((size_t)nw * NTOK + rowc[e]) * NTOK + c0];
      }
      if (c1 < NTOK) {
        const int idx = (rz[e] - c1z + 6) * 20 + (ry[e] - c1y + 6) * 13 + (rx[e] - c1x + 6);
        v1 = s1[e] + rel_table[idx * NHEAD + h]
                   + attn_mask[((size_t)nw * NTOK + rowc[e]) * NTOK + c1];
      }
      const float rmax = half_max(fmaxf(v0, v1));
      const float mnew = fmaxf(mrun[e], rmax);
      const float scale = __expf(mrun[e] - mnew);
      const float e0 = (c0 < NTOK) ? __expf(v0 - mnew) : 0.f;
      const float e1 = (c1 < NTOK) ? __expf(v1 - mnew) : 0.f;
      lrun[e] = lrun[e] * scale + half_sum(e0 + e1);
      o0[e] *= scale; o1[e] *= scale;
      mrun[e] = mnew;
      p0[e] = e0; p1[e] = e1;
    }
    // stage probs tile (16 rows x 32 keys) into LDS: C-layout -> A-fragment re-shape
#pragma unroll
    for (int e = 0; e < 8; ++e) {
      const int rl = e + 8 * half;
      pl[rl * 32 + r]      = (bf16_t)p0[e];
      pl[rl * 32 + 16 + r] = (bf16_t)p1[e];
    }
    asm volatile("s_wait_dscnt 0x0" ::: "memory");  // DS pipe in-order per wave
    const v16bf ap  = load_frag_bf16(pl, 32, r, 0, half);
    const v16bf bv0 = load_frag_bf16(lds_v, 32, r, 0, half);
    const v16bf bv1 = load_frag_bf16(lds_v, 32, 16 + r, 0, half);
    o0 = __builtin_amdgcn_wmma_f32_16x16x32_bf16(false, ap, false, bv0, (short)0, o0, false, false);
    o1 = __builtin_amdgcn_wmma_f32_16x16x32_bf16(false, ap, false, bv1, (short)0, o1, false, false);
  }

  if (active) {
#pragma unroll
    for (int e = 0; e < 8; ++e) {
      const int row = tm * 16 + e + 8 * half;
      const float inv = 1.f / lrun[e];
      bf16_t* dst = ob + ((size_t)win * NPAD + row) * DIMC + h * HD;
      dst[r]      = (bf16_t)(o0[e] * inv);
      dst[16 + r] = (bf16_t)(o1[e] * inv);
    }
  }
}

// ---------------- proj + window reverse + roll back + residual ----------------
__global__ void k_proj_gemm(const bf16_t* __restrict__ ob, const bf16_t* __restrict__ Bt,
                            const float* __restrict__ proj_b, const float* __restrict__ x,
                            float* __restrict__ xr) {
  const int wave = threadIdx.x >> 5;
  const int tile = blockIdx.x * 8 + wave;
  const int TM = NPAD / 32, TN = DIMC / 32;  // 11 x 6
  if (tile >= TM * TN) return;
  const int win = blockIdx.y;
  const int tm = tile / TN, tn = tile % TN;
  v8f acc[2][2] = {{zero_acc(), zero_acc()}, {zero_acc(), zero_acc()}};
  gemm_block_bf16<2, 2>(ob + (size_t)win * NPAD * DIMC, DIMC, Bt, DIMC, DIMC,
                        tm * 32, tn * 32, acc);
  const int lane = threadIdx.x & 31, half = lane >> 4, r = lane & 15;
  const int b = win >> 6, w = win & 63;
  const int sz = w >> 4, sy = (w >> 2) & 3, sx = w & 3;
#pragma unroll
  for (int j = 0; j < 2; ++j) {
    const int col = tn * 32 + j * 16 + r;
    const float pb = proj_b[col];
#pragma unroll
    for (int i = 0; i < 2; ++i)
#pragma unroll
      for (int e = 0; e < 8; ++e) {
        const int row = tm * 32 + i * 16 + e + 8 * half;
        if (row >= NTOK) continue;
        const int zi = row / 49, yi = (row / 7) % 7, xi = row % 7;
        const int z  = (sz * 7 + zi + 3) % 28;   // roll(+3) == inverse of roll(-3)
        const int y  = (sy * 7 + yi + 3) % 28;
        const int xx = (sx * 7 + xi + 3) % 28;
        const size_t idx = ((size_t)b * SEQ + (z * 28 + y) * 28 + xx) * DIMC + col;
        xr[idx] = x[idx] + acc[i][j][e] + pb;    // residual 1
      }
  }
}

// ---------------- LN2 -> bf16 A for MLP ----------------
__global__ void k_ln2(const float* __restrict__ xr, const float* __restrict__ g,
                      const float* __restrict__ bb, bf16_t* __restrict__ A2) {
  const int wave = threadIdx.x >> 5, lane = threadIdx.x & 31;
  const int row = blockIdx.x * 8 + wave;
  if (row >= MROWS) return;
  const float* src = xr + (size_t)row * DIMC;
  bf16_t* dst = A2 + (size_t)row * DIMC;
  float v[6]; float s = 0.f;
#pragma unroll
  for (int i = 0; i < 6; ++i) { v[i] = src[lane + 32 * i]; s += v[i]; }
  const float mean = wave_sum(s) * (1.f / DIMC);
  float q = 0.f;
#pragma unroll
  for (int i = 0; i < 6; ++i) { const float d = v[i] - mean; q += d * d; }
  const float rstd = rsqrtf(wave_sum(q) * (1.f / DIMC) + 1e-5f);
#pragma unroll
  for (int i = 0; i < 6; ++i) {
    const int c = lane + 32 * i;
    dst[c] = (bf16_t)((v[i] - mean) * rstd * g[c] + bb[c]);
  }
}

// ---------------- fc1 + exact GELU ----------------
__global__ void k_fc1_gemm(const bf16_t* __restrict__ A2, const bf16_t* __restrict__ Bt,
                           const float* __restrict__ b1, bf16_t* __restrict__ h1) {
  const int wave = threadIdx.x >> 5;
  const int tile = blockIdx.x * 8 + wave;
  const int TN = FFN / 32;                   // 24, TM = 1372
  if (tile >= (MROWS / 32) * TN) return;
  const int tm = tile / TN, tn = tile % TN;
  v8f acc[2][2] = {{zero_acc(), zero_acc()}, {zero_acc(), zero_acc()}};
  gemm_block_bf16<2, 2>(A2, DIMC, Bt, DIMC, DIMC, tm * 32, tn * 32, acc);
  const int lane = threadIdx.x & 31, half = lane >> 4, r = lane & 15;
#pragma unroll
  for (int j = 0; j < 2; ++j) {
    const int col = tn * 32 + j * 16 + r;
    const float bv = b1[col];
#pragma unroll
    for (int i = 0; i < 2; ++i)
#pragma unroll
      for (int e = 0; e < 8; ++e) {
        const int row = tm * 32 + i * 16 + e + 8 * half;
        const float u = acc[i][j][e] + bv;
        h1[(size_t)row * FFN + col] =
            (bf16_t)(0.5f * u * (1.f + erff(u * 0.70710678118654752f)));
      }
  }
}

// ---------------- fc2 + residual 2 -> d_out ----------------
__global__ void k_fc2_gemm(const bf16_t* __restrict__ h1, const bf16_t* __restrict__ Bt,
                           const float* __restrict__ b2, const float* __restrict__ xr,
                           float* __restrict__ out) {
  const int wave = threadIdx.x >> 5;
  const int tile = blockIdx.x * 8 + wave;
  const int TN = DIMC / 32;                  // 6, TM = 1372
  if (tile >= (MROWS / 32) * TN) return;
  const int tm = tile / TN, tn = tile % TN;
  v8f acc[2][2] = {{zero_acc(), zero_acc()}, {zero_acc(), zero_acc()}};
  gemm_block_bf16<2, 2>(h1, FFN, Bt, FFN, FFN, tm * 32, tn * 32, acc);
  const int lane = threadIdx.x & 31, half = lane >> 4, r = lane & 15;
#pragma unroll
  for (int j = 0; j < 2; ++j) {
    const int col = tn * 32 + j * 16 + r;
    const float bv = b2[col];
#pragma unroll
    for (int i = 0; i < 2; ++i)
#pragma unroll
      for (int e = 0; e < 8; ++e) {
        const int row = tm * 32 + i * 16 + e + 8 * half;
        const size_t idx = (size_t)row * DIMC + col;
        out[idx] = xr[idx] + acc[i][j][e] + bv;
      }
  }
}

// ---------------- launch ----------------
extern "C" void kernel_launch(void* const* d_in, const int* in_sizes, int n_in,
                              void* d_out, int out_size, void* d_ws, size_t ws_size,
                              hipStream_t stream) {
  const float* x         = (const float*)d_in[0];
  const float* attn_mask = (const float*)d_in[1];
  const float* norm1_g   = (const float*)d_in[2];
  const float* norm1_b   = (const float*)d_in[3];
  const float* qkv_w     = (const float*)d_in[4];
  const float* qkv_b     = (const float*)d_in[5];
  const float* rel_table = (const float*)d_in[6];
  const float* proj_w    = (const float*)d_in[7];
  const float* proj_b    = (const float*)d_in[8];
  const float* norm2_g   = (const float*)d_in[9];
  const float* norm2_b   = (const float*)d_in[10];
  const float* fc1_w     = (const float*)d_in[11];
  const float* fc1_b     = (const float*)d_in[12];
  const float* fc2_w     = (const float*)d_in[13];
  const float* fc2_b     = (const float*)d_in[14];
  float* out = (float*)d_out;

  char* ws = (char*)d_ws;
  size_t off = 0;
  auto alloc = [&](size_t bytes) -> void* {
    void* p = ws + off;
    off += (bytes + 255) & ~(size_t)255;
    return p;
  };
  bf16_t* Wbuf   = (bf16_t*)alloc((size_t)NWIN * NPAD * DIMC * 2);
  bf16_t* qb     = (bf16_t*)alloc((size_t)NWIN * NHEAD * NPAD * HD * 2);
  bf16_t* kb     = (bf16_t*)alloc((size_t)NWIN * NHEAD * NPAD * HD * 2);
  bf16_t* vT     = (bf16_t*)alloc((size_t)NWIN * NHEAD * HD * NPAD * 2);
  bf16_t* ob     = (bf16_t*)alloc((size_t)NWIN * NPAD * DIMC * 2);
  float*  xr     = (float*) alloc((size_t)MROWS * DIMC * 4);
  bf16_t* A2     = (bf16_t*)alloc((size_t)MROWS * DIMC * 2);
  bf16_t* h1     = (bf16_t*)alloc((size_t)MROWS * FFN * 2);
  bf16_t* qkvWt  = (bf16_t*)alloc((size_t)576 * DIMC * 2);
  bf16_t* projWt = (bf16_t*)alloc((size_t)DIMC * DIMC * 2);
  bf16_t* fc1Wt  = (bf16_t*)alloc((size_t)FFN * DIMC * 2);
  bf16_t* fc2Wt  = (bf16_t*)alloc((size_t)DIMC * FFN * 2);

  const dim3 blk(256);
  k_transpose_w<<<(DIMC * 576 + 255) / 256, blk, 0, stream>>>(qkv_w, qkvWt, DIMC, 576);
  k_transpose_w<<<(DIMC * DIMC + 255) / 256, blk, 0, stream>>>(proj_w, projWt, DIMC, DIMC);
  k_transpose_w<<<(DIMC * FFN + 255) / 256, blk, 0, stream>>>(fc1_w, fc1Wt, DIMC, FFN);
  k_transpose_w<<<(FFN * DIMC + 255) / 256, blk, 0, stream>>>(fc2_w, fc2Wt, FFN, DIMC);

  k_ln1_shift_partition<<<dim3(NPAD / 8, NWIN), blk, 0, stream>>>(x, norm1_g, norm1_b, Wbuf);
  k_qkv_gemm  <<<dim3(25, NWIN),         blk, 0, stream>>>(Wbuf, qkvWt, qkv_b, qb, kb, vT);
  k_flash_attn<<<dim3(3, NWIN * NHEAD),  blk, 0, stream>>>(qb, kb, vT, rel_table, attn_mask, ob);
  k_proj_gemm <<<dim3(9, NWIN),          blk, 0, stream>>>(ob, projWt, proj_b, x, xr);
  k_ln2       <<<dim3(MROWS / 8, 1),     blk, 0, stream>>>(xr, norm2_g, norm2_b, A2);
  k_fc1_gemm  <<<dim3((MROWS / 32) * (FFN  / 32) / 8, 1), blk, 0, stream>>>(A2, fc1Wt, fc1_b, h1);
  k_fc2_gemm  <<<dim3((MROWS / 32) * (DIMC / 32) / 8, 1), blk, 0, stream>>>(h1, fc2Wt, fc2_b, xr, out);
}
